// VectorQuantizer_71614284693864
// MI455X (gfx1250) — matrix operand order repaired and checked by reading, not compiled, not run on hardware
//
#include <hip/hip_runtime.h>
#include <hip/hip_bf16.h>
#include <math.h>

// ---------------------------------------------------------------------------
// VQ-VAE quantizer, MI455X (gfx1250, wave32, WMMA + async global->LDS).
//   z:         (16, 256, 32, 32) f32     -> 16384 rows x 256 dims
//   embedding: (16384, 256) f32          -> 16384 codes x 256 dims
// Outputs (flat, concat): z_q (4194304 f32) | loss (1) | perplexity (1) |
//                         min_encoding (16384*16384 f32) | idx (16384 i32)
// ---------------------------------------------------------------------------

typedef __attribute__((ext_vector_type(16))) __bf16 v16bf;
typedef __attribute__((ext_vector_type(8)))  __bf16 v8bf;
typedef __attribute__((ext_vector_type(8)))  float  v8f;

#define NROWS 16384
#define NCODE 16384
#define KDIM  256
#define NCHUNK 8              // 256 / 32
#define NTILES_COL 1024       // 16384 / 16
#define NTILES_ROW 1024
#define TILE_BYTES 8192       // one B tile: 8 chunks * 32 lanes * 32B

// ---- CDNA5 async global->LDS copy (ASYNCcnt) -------------------------------
__device__ __forceinline__ void async_copy_b128(unsigned lds_off, const void* gsrc) {
    // VDST = per-lane LDS byte offset, VADDR = 64-bit global address
    asm volatile("global_load_async_to_lds_b128 %0, %1, off"
                 :: "v"(lds_off), "v"(gsrc) : "memory");
}

__device__ __forceinline__ void wait_async0() {
#if __has_builtin(__builtin_amdgcn_s_wait_asynccnt)
    __builtin_amdgcn_s_wait_asynccnt(0);
#else
    asm volatile("s_wait_asynccnt 0" ::: "memory");
#endif
}

// ---------------- zero fills ------------------------------------------------
__global__ void vq_zero_enc_kernel(float2* __restrict__ enc, size_t count2) {
    size_t stride = (size_t)gridDim.x * blockDim.x;
    for (size_t i = (size_t)blockIdx.x * blockDim.x + threadIdx.x; i < count2; i += stride) {
        float2 zz; zz.x = 0.0f; zz.y = 0.0f;
        enc[i] = zz;
    }
}

__global__ void vq_zero_hist_kernel(int* __restrict__ hist) {
    hist[blockIdx.x * blockDim.x + threadIdx.x] = 0;
}

// ---------------- ||e||^2 ---------------------------------------------------
__global__ void vq_enorm_kernel(const float* __restrict__ e, float* __restrict__ enorm) {
    int code = blockIdx.x * blockDim.x + threadIdx.x;
    const float* er = e + (size_t)code * KDIM;
    float s = 0.0f;
#pragma unroll 8
    for (int i = 0; i < KDIM; ++i) { float v = er[i]; s += v * v; }
    enorm[code] = s;
}

// ---------------- pack A (z rows) into WMMA 16x32 bf16 A-fragments ----------
// ISA 7.12.2, 16-bit A 16x32: lanes 0-15 M=lane, V0..3 K=0..7, V4..7 K=16..23;
// lanes 16-31 same M, V0..3 K=8..15, V4..7 K=24..31.
__global__ void vq_pack_a_kernel(const float* __restrict__ z, __bf16* __restrict__ apack) {
    int tile  = blockIdx.x;
    int chunk = threadIdx.x >> 5;
    int lane  = threadIdx.x & 31;
    int hf    = lane >> 4;
    int m     = lane & 15;
    int n  = tile * 16 + m;          // flat (b,h,w) row
    int b  = n >> 10;
    int hw = n & 1023;
    const float* zb = z + (size_t)b * (KDIM * 1024) + hw;   // channel stride = 1024
    v16bf frag;
#pragma unroll
    for (int j = 0; j < 16; ++j) {
        int v = j >> 1;
        int k = ((v < 4) ? 0 : 16) + (hf ? 8 : 0) + ((v & 3) << 1) + (j & 1);
        int c = chunk * 32 + k;
        frag[j] = (__bf16)zb[(size_t)c * 1024];
    }
    ((v16bf*)apack)[(tile * NCHUNK + chunk) * 32 + lane] = frag;
}

// ---------------- pack B (codebook cols) into WMMA 32x16 bf16 B-fragments ---
// B 32x16: lanes 0-15 N=lane hold K=0..7 (V0..V3), K=16..23 (V4..V7);
// lanes 16-31 N=lane-16 hold K=8..15, K=24..31.
// Storage is split: elements j=0..7 (V0..V3) in the tile's first 4KB,
// j=8..15 (V4..V7) in the second 4KB, each at lane-stride 16B so the
// consumer's ds_load_b128 is bank-conflict-free.
__global__ void vq_pack_b_kernel(const float* __restrict__ e, __bf16* __restrict__ bpack) {
    int tile  = blockIdx.x;
    int chunk = threadIdx.x >> 5;
    int lane  = threadIdx.x & 31;
    int hf    = lane >> 4;
    int nn    = lane & 15;
    int code  = tile * 16 + nn;
    const float* er = e + (size_t)code * KDIM + chunk * 32 + hf * 8;
    v8bf lo, hi;
#pragma unroll
    for (int j = 0; j < 8; ++j)  lo[j] = (__bf16)er[j];
#pragma unroll
    for (int j = 0; j < 8; ++j)  hi[j] = (__bf16)er[16 + j];
    v8bf* dst = (v8bf*)(bpack + (size_t)tile * (TILE_BYTES / 2));  // 4096 elems/tile
    dst[chunk * 32 + lane]       = lo;   // first 4KB half
    dst[256 + chunk * 32 + lane] = hi;   // second 4KB half
}

// ---------------- fused GEMM + argmin (the hot loop) ------------------------
// One wave owns 16 z-rows (A panel resident in 64 VGPRs). The 8 waves of a
// block share each 8KB B tile: staged once per block into LDS with
// global_load_async_to_lds_b128 (double-buffered), consumed via ds_load_b128,
// 8 x v_wmma_f32_16x16x32_bf16 per tile, then running min of
// d = ||e||^2 - 2*z.e on the C fragment (||z||^2 is row-constant).
__global__ void __launch_bounds__(256) vq_argmin_wmma_kernel(
    const __bf16* __restrict__ apack, const __bf16* __restrict__ bpack,
    const float* __restrict__ enorm, int* __restrict__ mini) {
    __shared__ __align__(16) unsigned char smem[2 * TILE_BYTES];

    const int tid  = threadIdx.x;
    const int wave = tid >> 5;
    const int lane = tid & 31;
    const int hf   = lane >> 4;
    const int nlo  = lane & 15;
    const int rowTile = blockIdx.x * 8 + wave;

    const v16bf* ap = (const v16bf*)apack;
    v16bf a[NCHUNK];
#pragma unroll
    for (int c = 0; c < NCHUNK; ++c)
        a[c] = ap[(rowTile * NCHUNK + c) * 32 + lane];

    float minv[8];
    int   minc[8];
#pragma unroll
    for (int r = 0; r < 8; ++r) { minv[r] = 3.4e38f; minc[r] = 0; }

    const unsigned char* gsrc = (const unsigned char*)bpack;
    const unsigned lds_base = (unsigned)(unsigned long long)(const void*)smem;

    // prologue: stage tile 0 into buffer 0 (each thread moves 2 x 16B)
    async_copy_b128(lds_base + tid * 16,        gsrc + tid * 16);
    async_copy_b128(lds_base + tid * 16 + 4096, gsrc + tid * 16 + 4096);
    wait_async0();
    __syncthreads();

    for (int ct = 0; ct < NTILES_COL; ++ct) {
        const int cur = ct & 1;
        const int nxt = cur ^ 1;

        if (ct + 1 < NTILES_COL) {      // uniform branch: stage next tile
            const unsigned char* s = gsrc + (size_t)(ct + 1) * TILE_BYTES + tid * 16;
            unsigned d = lds_base + nxt * TILE_BYTES + tid * 16;
            async_copy_b128(d,        s);
            async_copy_b128(d + 4096, s + 4096);
        }
        if (ct + 4 < NTILES_COL) {      // warm L2 ahead of the async stream
            __builtin_prefetch(gsrc + (size_t)(ct + 4) * TILE_BYTES + tid * 16, 0, 1);
            __builtin_prefetch(gsrc + (size_t)(ct + 4) * TILE_BYTES + 4096 + tid * 16, 0, 1);
        }

        float en = enorm[ct * 16 + nlo];
        v8f acc = {};
        const v8bf* lbuf = (const v8bf*)(smem + cur * TILE_BYTES);
#pragma unroll
        for (int c = 0; c < NCHUNK; ++c) {
            v8bf lo = lbuf[c * 32 + lane];          // ds_load_b128, conflict-free
            v8bf hi = lbuf[256 + c * 32 + lane];    // ds_load_b128, conflict-free
            v16bf bfrag = __builtin_shufflevector(lo, hi,
                0, 1, 2, 3, 4, 5, 6, 7, 8, 9, 10, 11, 12, 13, 14, 15);
            acc = __builtin_amdgcn_wmma_f32_16x16x32_bf16(
                false, a[c], false, bfrag, (short)0, acc, false, false);
        }

        int col = ct * 16 + nlo;
#pragma unroll
        for (int r = 0; r < 8; ++r) {
            float d = en - 2.0f * acc[r];
            if (d < minv[r]) { minv[r] = d; minc[r] = col; }
        }

        wait_async0();                  // our part of tile ct+1 has landed
        __syncthreads();                // everyone's part has landed
    }

    // C layout: VGPR r, lanes 0-15 -> row r; lanes 16-31 -> row r+8.
    // Min-reduce across the 16 lanes of each half (prefer lower index on ties).
#pragma unroll
    for (int r = 0; r < 8; ++r) {
        float v = minv[r];
        int   i = minc[r];
#pragma unroll
        for (int off = 8; off > 0; off >>= 1) {
            float ov = __shfl_xor(v, off, 16);
            int   oi = __shfl_xor(i, off, 16);
            if (ov < v || (ov == v && oi < i)) { v = ov; i = oi; }
        }
        if (nlo == 0) {
            int row = rowTile * 16 + r + 8 * hf;
            mini[row] = i;
        }
    }
}

// ---------------- scatter: idx out, one-hot ones, histogram -----------------
__global__ void vq_scatter_kernel(const int* __restrict__ mini,
                                  float* __restrict__ enc,
                                  int* __restrict__ idx_out,
                                  int* __restrict__ hist) {
    int n = blockIdx.x * blockDim.x + threadIdx.x;
    int code = mini[n];
    idx_out[n] = code;                             // int32 bit pattern
    enc[(size_t)n * NCODE + code] = 1.0f;
    atomicAdd(&hist[code], 1);                     // integer: deterministic
}

// ---------------- z_q gather (output-coalesced) + squared-diff partials -----
__global__ void vq_zq_loss_kernel(const float* __restrict__ z,
                                  const float* __restrict__ e,
                                  const int* __restrict__ mini,
                                  float* __restrict__ zq_out,
                                  float* __restrict__ partial_sq) {
    size_t o = (size_t)blockIdx.x * blockDim.x + threadIdx.x; // 0..4194303
    int b  = (int)(o >> 18);
    int c  = (int)(o >> 10) & 255;
    int hw = (int)o & 1023;
    int n  = (b << 10) + hw;
    int code = mini[n];
    float q  = e[(size_t)code * KDIM + c];
    float zz = z[o];
    zq_out[o] = q;                                 // straight-through == z_q
    float d = q - zz;

    __shared__ float sm[256];
    sm[threadIdx.x] = d * d;
    __syncthreads();
    for (int s = 128; s > 0; s >>= 1) {
        if (threadIdx.x < s) sm[threadIdx.x] += sm[threadIdx.x + s];
        __syncthreads();
    }
    if (threadIdx.x == 0) partial_sq[blockIdx.x] = sm[0];
}

// ---------------- entropy partials over histogram ---------------------------
__global__ void vq_entropy_kernel(const int* __restrict__ hist,
                                  float* __restrict__ partial_ent) {
    int i = blockIdx.x * blockDim.x + threadIdx.x;
    float p = (float)hist[i] * (1.0f / (float)NROWS);
    float t = p * logf(p + 1e-10f);
    __shared__ float sm[256];
    sm[threadIdx.x] = t;
    __syncthreads();
    for (int s = 128; s > 0; s >>= 1) {
        if (threadIdx.x < s) sm[threadIdx.x] += sm[threadIdx.x + s];
        __syncthreads();
    }
    if (threadIdx.x == 0) partial_ent[blockIdx.x] = sm[0];
}

// ---------------- finalize: loss + perplexity scalars -----------------------
__global__ void vq_finalize_kernel(const float* __restrict__ partial_sq,
                                   const float* __restrict__ partial_ent,
                                   float* __restrict__ loss_out,
                                   float* __restrict__ perp_out) {
    __shared__ float sm[256];
    float s = 0.0f;
    for (int i = threadIdx.x; i < 16384; i += 256) s += partial_sq[i];
    sm[threadIdx.x] = s;
    __syncthreads();
    for (int st = 128; st > 0; st >>= 1) {
        if (threadIdx.x < st) sm[threadIdx.x] += sm[threadIdx.x + st];
        __syncthreads();
    }
    float total_sq = sm[0];
    __syncthreads();

    float s2 = 0.0f;
    for (int i = threadIdx.x; i < 64; i += 256) s2 += partial_ent[i];
    sm[threadIdx.x] = s2;
    __syncthreads();
    for (int st = 128; st > 0; st >>= 1) {
        if (threadIdx.x < st) sm[threadIdx.x] += sm[threadIdx.x + st];
        __syncthreads();
    }
    if (threadIdx.x == 0) {
        // loss = mean(diff^2) + BETA*mean(diff^2) in forward
        loss_out[0] = 1.25f * total_sq / 4194304.0f;
        perp_out[0] = expf(-sm[0]);
    }
}

// ---------------------------------------------------------------------------
extern "C" void kernel_launch(void* const* d_in, const int* in_sizes, int n_in,
                              void* d_out, int out_size, void* d_ws, size_t ws_size,
                              hipStream_t stream) {
    const float* z = (const float*)d_in[0];        // 16*256*32*32
    const float* e = (const float*)d_in[1];        // 16384*256
    float* out = (float*)d_out;

    char* ws = (char*)d_ws;
    __bf16* apack = (__bf16*)ws;                                    // 8 MB
    __bf16* bpack = (__bf16*)(ws + (size_t)(8u << 20));             // 8 MB
    float*  enorm = (float*)(ws + (size_t)(16u << 20));             // 64 KB
    int*    mini  = (int*)  (ws + (size_t)(16u << 20) + (64u << 10));
    int*    hist  = (int*)  (ws + (size_t)(16u << 20) + (128u << 10));
    float*  psq   = (float*)(ws + (size_t)(16u << 20) + (192u << 10)); // 64 KB
    float*  pent  = (float*)(ws + (size_t)(16u << 20) + (256u << 10)); // 256 B

    const size_t off_loss = 4194304;
    const size_t off_perp = 4194305;
    const size_t off_enc  = 4194306;
    const size_t off_idx  = off_enc + (size_t)NROWS * NCODE;

    vq_zero_hist_kernel<<<64, 256, 0, stream>>>(hist);
    vq_zero_enc_kernel<<<65536, 256, 0, stream>>>(
        (float2*)(out + off_enc), (size_t)NROWS * NCODE / 2);
    vq_enorm_kernel<<<64, 256, 0, stream>>>(e, enorm);
    vq_pack_a_kernel<<<NTILES_ROW, 256, 0, stream>>>(z, apack);
    vq_pack_b_kernel<<<NTILES_COL, 256, 0, stream>>>(e, bpack);
    vq_argmin_wmma_kernel<<<128, 256, 0, stream>>>(apack, bpack, enorm, mini);
    vq_scatter_kernel<<<64, 256, 0, stream>>>(mini, out + off_enc,
                                              (int*)(out + off_idx), hist);
    vq_zq_loss_kernel<<<16384, 256, 0, stream>>>(z, e, mini, out, psq);
    vq_entropy_kernel<<<64, 256, 0, stream>>>(hist, pent);
    vq_finalize_kernel<<<1, 256, 0, stream>>>(psq, pent,
                                              out + off_loss, out + off_perp);
}
